// RnnDecoder_19902878450233
// MI455X (gfx1250) — matrix operand order repaired and checked by reading, not compiled, not run on hardware
//
#include <hip/hip_runtime.h>

// ---------------------------------------------------------------------------
// LSTM decoder for gfx1250 (MI455X): bf16 WMMA GEMMs + fused f32 cell update.
//   IN_FEAT=512, ACT_FEAT=512, HIDDEN=1024, L=128, B=64
//   out = [hs (128*64*1024) | h_n (64*1024) | c_n (64*1024)]  (f32)
//
// Per step, ONE kernel: each block owns 32 hidden columns across all 4 gates
// (64x128 GEMM tile, K=1536 fused [h | act_t]). K is processed in 64-wide
// chunks through a double-buffered LDS pipeline:
//     global loads (chunk c+1 -> regs)  |  WMMAs on chunk c from LDS
// so L2 latency overlaps the matrix pipe even at ~2 waves/SIMD occupancy.
// Gates are then staged in (overlaid) LDS and the sigmoid/tanh cell update
// runs in-block. h ping-pongs between two bf16 buffers across launches.
// ---------------------------------------------------------------------------

typedef __attribute__((ext_vector_type(16))) __bf16 v16bf;
typedef __attribute__((ext_vector_type(8)))  float  v8f;

#define HID    1024
#define GATES  4096      // 4*HID
#define BATCH  64
#define LSTEPS 128
#define INFEAT 512
#define ACTF   512
#define KTOT   (HID + ACTF)   // 1536 fused K per step
#define CHUNK  64             // K elements staged per pipeline stage
#define NCHUNK (KTOT / CHUNK) // 24
#define NJ     32        // hidden columns per block in the fused step kernel
#define LDSP   132       // padded gate-row stride (floats) -> conflict-free
#define APAD   72        // padded LDS row (bf16) for A chunk (36 dwords, 16B-aligned)
#define BPAD   72        // padded LDS row (bf16) for B chunk

#define A_BUF_ELEMS (BATCH * APAD)   // 64*72
#define B_BUF_ELEMS (128 * BPAD)     // 128*72
#define SMEM_BYTES  (2 * A_BUF_ELEMS * 2 + 2 * B_BUF_ELEMS * 2)   // 55296

__device__ __forceinline__ unsigned short f2bf(float x) {
  unsigned u = __float_as_uint(x);
  unsigned r = u + 0x7FFFu + ((u >> 16) & 1u);   // round-to-nearest-even
  return (unsigned short)(r >> 16);
}

union FragAB { uint4 q[2]; unsigned u[8]; v16bf v; };

// ---- global-memory fragment loads (used by the one-shot base GEMM) --------
__device__ __forceinline__ v16bf load_a(const unsigned short* A, int row0, int k0,
                                        int lda, int lane) {
  FragAB f;
  int m  = lane & 15;
  int kb = k0 + ((lane >> 4) << 3);
  const unsigned* row = (const unsigned*)(A + (size_t)(row0 + m) * lda);
  int d = kb >> 1;
#pragma unroll
  for (int j = 0; j < 4; ++j) { f.u[j] = row[d + j]; f.u[4 + j] = row[d + 8 + j]; }
  return f.v;
}
__device__ __forceinline__ v16bf load_b(const unsigned short* W, int n0, int k0,
                                        int ldw, int lane) {
  FragAB f;
  int n  = lane & 15;
  int kb = k0 + ((lane >> 4) << 3);
  const unsigned* row = (const unsigned*)(W + (size_t)(n0 + n) * ldw);
  int d = kb >> 1;
#pragma unroll
  for (int v = 0; v < 4; ++v) { f.u[v] = row[d + v]; f.u[4 + v] = row[d + 8 + v]; }
  return f.v;
}

// ---- LDS fragment loads (padded rows, ds_load_b128 x2) --------------------
__device__ __forceinline__ v16bf lds_load_a(const unsigned short* As, int mstrip,
                                            int ks, int lane) {
  int m  = lane & 15;
  int kb = ks + ((lane >> 4) << 3);
  const unsigned* row = (const unsigned*)(As + (mstrip + m) * APAD);
  int d = kb >> 1;                       // 0,4,16,20 -> 16B aligned
  FragAB f;
  f.q[0] = *(const uint4*)(row + d);
  f.q[1] = *(const uint4*)(row + d + 8);
  return f.v;
}
__device__ __forceinline__ v16bf lds_load_b(const unsigned short* Bs, int nt,
                                            int ks, int lane) {
  int n  = lane & 15;
  int kb = ks + ((lane >> 4) << 3);
  const unsigned* row = (const unsigned*)(Bs + (nt * 16 + n) * BPAD);
  int d = kb >> 1;                       // 0,4,16,20 -> 16B aligned
  FragAB f;
  f.q[0] = *(const uint4*)(row + d);
  f.q[1] = *(const uint4*)(row + d + 8);
  return f.v;
}

// ---- chunk staging: global -> regs -> LDS ---------------------------------
// A chunk: 64 rows x 64 K (8 KB), 2 x uint4 per thread.
// B chunk: 128 rows x 64 K (16 KB), 4 x uint4 per thread.
struct Stage { uint4 a[2]; uint4 b[4]; };

__device__ __forceinline__ void stage_load(Stage& s, int tid, int kk, int j0,
                                           const unsigned short* hin,
                                           const unsigned short* act_t,
                                           const unsigned short* whh,
                                           const unsigned short* wih) {
  const unsigned short* asrc; int alda, acol;
  if (kk < HID) { asrc = hin;   alda = HID;  acol = kk; }
  else          { asrc = act_t; alda = ACTF; acol = kk - HID; }
  int arow = tid >> 2;
  int aseg = (tid & 3) * 2;                       // uint4 index within row
  const uint4* ap = (const uint4*)(asrc + (size_t)arow * alda + acol);
  s.a[0] = ap[aseg];
  s.a[1] = ap[aseg + 1];

  const unsigned short* bsrc; int bcol;
  if (kk < HID) { bsrc = whh; bcol = kk; }
  else          { bsrc = wih; bcol = INFEAT + (kk - HID); }
  int brow = tid >> 1;                            // local B row 0..127
  int g = (brow >> 5) * HID + j0 + (brow & 31);   // gate column (W row)
  const uint4* bp = (const uint4*)(bsrc + (size_t)g * HID + bcol);
  int bseg = (tid & 1) * 4;
#pragma unroll
  for (int q = 0; q < 4; ++q) s.b[q] = bp[bseg + q];
}

__device__ __forceinline__ void stage_store(const Stage& s, int tid,
                                            unsigned short* As, unsigned short* Bs) {
  int arow = tid >> 2;
  int aseg = (tid & 3) * 2;
  uint4* ad = (uint4*)((unsigned*)As + arow * (APAD / 2) + aseg * 4);
  ad[0] = s.a[0];
  ad[1] = s.a[1];

  int brow = tid >> 1;
  int bseg = (tid & 1) * 4;
  uint4* bd = (uint4*)((unsigned*)Bs + brow * (BPAD / 2) + bseg * 4);
#pragma unroll
  for (int q = 0; q < 4; ++q) bd[q] = s.b[q];
}

// ---------------------------------------------------------------------------
// f32 -> bf16 bulk conversion (grid-stride)
__global__ void cvt_bf16(const float* __restrict__ src, unsigned short* __restrict__ dst, int n) {
  int i = blockIdx.x * blockDim.x + threadIdx.x;
  int stride = gridDim.x * blockDim.x;
  for (; i < n; i += stride) dst[i] = f2bf(src[i]);
}

// initialize recurrent state from h0/c0 inputs
__global__ void init_state(const float* __restrict__ h0, const float* __restrict__ c0,
                           unsigned short* __restrict__ hb, float* __restrict__ c) {
  int i = blockIdx.x * blockDim.x + threadIdx.x;   // 65536 threads
  hb[i] = f2bf(h0[i]);
  c[i]  = c0[i];
}

// ---------------------------------------------------------------------------
// base[b][g] = input_bf16 . W_x^T + b_ih + b_hh   (one-shot, K=512)
// grid (8,4), block 256: wave = 16(M) x 64(N) strip.
__global__ void base_gemm(const unsigned short* __restrict__ inb,
                          const unsigned short* __restrict__ wih,
                          const float* __restrict__ b_ih, const float* __restrict__ b_hh,
                          float* __restrict__ base) {
  int lane = threadIdx.x & 31;
  int wave = threadIdx.x >> 5;
  int n0 = blockIdx.x * 512 + wave * 64;
  int m0 = blockIdx.y * 16;

  v8f acc[4];
#pragma unroll
  for (int t = 0; t < 4; ++t)
#pragma unroll
    for (int r = 0; r < 8; ++r) acc[t][r] = 0.0f;

  for (int k = 0; k < INFEAT; k += 32) {
    v16bf a = load_a(inb, m0, k, INFEAT, lane);
#pragma unroll
    for (int t = 0; t < 4; ++t) {
      v16bf b = load_b(wih, n0 + t * 16, k, HID, lane);
      acc[t] = __builtin_amdgcn_wmma_f32_16x16x32_bf16(false, a, false, b,
                                                       (short)0, acc[t], false, false);
    }
  }
  int mrow = m0 + ((lane >> 4) << 3);
  int col  = lane & 15;
#pragma unroll
  for (int t = 0; t < 4; ++t) {
    int g = n0 + t * 16 + col;
    float bias = b_ih[g] + b_hh[g];
#pragma unroll
    for (int r = 0; r < 8; ++r)
      base[(size_t)(mrow + r) * GATES + g] = acc[t][r] + bias;
  }
}

// ---------------------------------------------------------------------------
// Fused per-step kernel. Grid: 32 blocks x 256 threads (8 waves).
// Block j0 computes gates[0..63][{q*1024 + j0 .. +31} for q=0..3]  (64x128).
__global__ void __launch_bounds__(256)
step_fused(const unsigned short* __restrict__ hin,   // 64x1024 bf16
           const unsigned short* __restrict__ whh,   // 4096x1024 bf16
           const unsigned short* __restrict__ act_t, // 64x512 bf16 (this step)
           const unsigned short* __restrict__ wih,   // 4096x1024 (cols 512.. = W_a)
           const float* __restrict__ base,           // 64x4096
           float* __restrict__ c,                    // 64x1024 f32 state
           unsigned short* __restrict__ hout,        // 64x1024 bf16 (next step)
           float* __restrict__ hs_t,                 // 64x1024 f32 output slice
           float* __restrict__ hn, float* __restrict__ cn, int last) {
  __shared__ __align__(16) unsigned char smem[SMEM_BYTES];
  unsigned short* As = (unsigned short*)smem;                          // [2][64*APAD]
  unsigned short* Bs = (unsigned short*)(smem + 2 * A_BUF_ELEMS * 2);  // [2][128*BPAD]
  float* glds = (float*)smem;                                          // overlaid gates

  int tid  = threadIdx.x;
  int lane = tid & 31;
  int wave = tid >> 5;
  int mstrip = (wave >> 1) << 4;        // 0,16,32,48 : M rows of this wave
  int ngrp   = (wave & 1) << 2;         // 0 or 4     : first of 4 N-tiles
  int j0 = blockIdx.x * NJ;

  // Warm next step's action tile toward L2 while this GEMM runs (speculative).
  {
    const unsigned short* nxt = act_t + (size_t)BATCH * ACTF;
    int lin = blockIdx.x * blockDim.x + tid;                     // 0..8191
    __builtin_prefetch(nxt + ((size_t)lin << 2), 0, 1);
  }

  v8f acc[4];
#pragma unroll
  for (int t = 0; t < 4; ++t)
#pragma unroll
    for (int r = 0; r < 8; ++r) acc[t][r] = 0.0f;

  // ---- prologue: stage chunk 0 into buffer 0 ----
  Stage st;
  stage_load(st, tid, 0, j0, hin, act_t, whh, wih);
  stage_store(st, tid, As, Bs);
  __syncthreads();

  // ---- double-buffered pipeline over 24 chunks ----
  for (int cch = 0; cch < NCHUNK; ++cch) {
    int cur = cch & 1;
    unsigned short* Ac = As + cur * A_BUF_ELEMS;
    unsigned short* Bc = Bs + cur * B_BUF_ELEMS;
    bool more = (cch + 1 < NCHUNK);

    // 1) issue next chunk's global loads (no wait yet)
    if (more) stage_load(st, tid, (cch + 1) * CHUNK, j0, hin, act_t, whh, wih);

    // 2) WMMAs on current chunk straight from LDS (independent of fresh loads)
#pragma unroll
    for (int ks = 0; ks < CHUNK; ks += 32) {
      v16bf a = lds_load_a(Ac, mstrip, ks, lane);
#pragma unroll
      for (int t = 0; t < 4; ++t) {
        v16bf b = lds_load_b(Bc, ngrp + t, ks, lane);
        acc[t] = __builtin_amdgcn_wmma_f32_16x16x32_bf16(false, a, false, b,
                                                         (short)0, acc[t], false, false);
      }
    }

    // 3) commit staged registers to the other buffer (load-wait lands here)
    if (more) stage_store(st, tid, As + (cur ^ 1) * A_BUF_ELEMS,
                          Bs + (cur ^ 1) * B_BUF_ELEMS);
    __syncthreads();
  }

  // ---- gates (+ base) -> overlaid LDS: local col = nt*16 + col (== 32q+jj) --
  int mrow = mstrip + ((lane >> 4) << 3);
  int col  = lane & 15;
#pragma unroll
  for (int t = 0; t < 4; ++t) {
    int nt = ngrp + t;
    int g  = (nt >> 1) * HID + j0 + ((nt & 1) << 4) + col;
#pragma unroll
    for (int r = 0; r < 8; ++r)
      glds[(mrow + r) * LDSP + nt * 16 + col] =
          acc[t][r] + base[(size_t)(mrow + r) * GATES + g];
  }
  __syncthreads();

  // ---- in-block LSTM cell update for 64 rows x 32 hidden columns ----
  for (int e = tid; e < BATCH * NJ; e += blockDim.x) {
    int b  = e >> 5;
    int jj = e & 31;
    const float* row = glds + b * LDSP;
    float xi = row[jj];
    float xf = row[32 + jj];
    float xg = row[64 + jj];
    float xo = row[96 + jj];
    float i = 1.0f / (1.0f + __expf(-xi));
    float f = 1.0f / (1.0f + __expf(-xf));
    float gg = tanhf(xg);
    float o = 1.0f / (1.0f + __expf(-xo));
    int cidx = b * HID + j0 + jj;
    float cc = f * c[cidx] + i * gg;
    float h  = o * tanhf(cc);
    c[cidx]   = cc;
    hout[cidx] = f2bf(h);
    hs_t[cidx] = h;
    if (last) { hn[cidx] = h; cn[cidx] = cc; }
  }
}

// ---------------------------------------------------------------------------
extern "C" void kernel_launch(void* const* d_in, const int* in_sizes, int n_in,
                              void* d_out, int out_size, void* d_ws, size_t ws_size,
                              hipStream_t stream) {
  const float* input = (const float*)d_in[0];   // (64,512)
  const float* acts  = (const float*)d_in[1];   // (128,64,512)
  const float* h0    = (const float*)d_in[2];   // (64,1024)
  const float* c0    = (const float*)d_in[3];   // (64,1024)
  const float* W_ih  = (const float*)d_in[4];   // (4096,1024)
  const float* W_hh  = (const float*)d_in[5];   // (4096,1024)
  const float* b_ih  = (const float*)d_in[6];   // (4096,)
  const float* b_hh  = (const float*)d_in[7];   // (4096,)

  char* ws = (char*)d_ws;
  size_t off = 0;
  auto alloc = [&](size_t bytes) -> char* {
    char* p = ws + off;
    off += (bytes + 255) & ~(size_t)255;
    return p;
  };
  unsigned short* wih_b  = (unsigned short*)alloc((size_t)GATES * HID * 2);          // 8 MB
  unsigned short* whh_b  = (unsigned short*)alloc((size_t)GATES * HID * 2);          // 8 MB
  unsigned short* acts_b = (unsigned short*)alloc((size_t)LSTEPS * BATCH * ACTF * 2); // 8 MB
  unsigned short* in_b   = (unsigned short*)alloc((size_t)BATCH * INFEAT * 2);
  unsigned short* h_ping = (unsigned short*)alloc((size_t)BATCH * HID * 2);
  unsigned short* h_pong = (unsigned short*)alloc((size_t)BATCH * HID * 2);
  float* c_f  = (float*)alloc((size_t)BATCH * HID * 4);
  float* base = (float*)alloc((size_t)BATCH * GATES * 4);                            // 1 MB

  // one-time f32 -> bf16 conversions + state init
  cvt_bf16<<<2048, 256, 0, stream>>>(W_ih, wih_b, GATES * HID);
  cvt_bf16<<<2048, 256, 0, stream>>>(W_hh, whh_b, GATES * HID);
  cvt_bf16<<<2048, 256, 0, stream>>>(acts, acts_b, LSTEPS * BATCH * ACTF);
  cvt_bf16<<<128, 256, 0, stream>>>(input, in_b, BATCH * INFEAT);
  init_state<<<(BATCH * HID) / 256, 256, 0, stream>>>(h0, c0, h_ping, c_f);

  // base = input @ W_x^T + b_ih + b_hh   (broadcast over all steps)
  base_gemm<<<dim3(8, 4), 256, 0, stream>>>(in_b, wih_b, b_ih, b_hh, base);

  float* hs = (float*)d_out;                       // (128,64,1024)
  float* hn = hs + (size_t)LSTEPS * BATCH * HID;   // (64,1024)
  float* cn = hn + (size_t)BATCH * HID;            // (64,1024)

  unsigned short* hbuf[2] = {h_ping, h_pong};
  for (int t = 0; t < LSTEPS; ++t) {
    step_fused<<<HID / NJ, 256, 0, stream>>>(
        hbuf[t & 1], whh_b, acts_b + (size_t)t * BATCH * ACTF, wih_b, base,
        c_f, hbuf[(t + 1) & 1], hs + (size_t)t * BATCH * HID, hn, cn,
        t == LSTEPS - 1);
  }
}
